// Net_31834297598336
// MI455X (gfx1250) — matrix operand order repaired and checked by reading, not compiled, not run on hardware
//
#include <hip/hip_runtime.h>
#include <math.h>

// ---------------------------------------------------------------- types
typedef __bf16 bf16;
typedef bf16  v16bf __attribute__((ext_vector_type(16)));
typedef bf16  v8bf  __attribute__((ext_vector_type(8)));
typedef float v8f   __attribute__((ext_vector_type(8)));

// ---------------------------------------------------------------- shapes
#define B_    32
#define T_    512
#define D_    768
#define HID   409
#define HP    416          // hidden padded to 13*32
#define GP    1664         // 4*HP (gate dim padded)
#define KP    832          // layer input width padded (818 -> 26*32)
#define TB    (T_ * B_)    // 16384 rows
#define NL_   4
#define VOC   17

#define WMMA_BF16(a, b, c) \
  __builtin_amdgcn_wmma_f32_16x16x32_bf16(false, (a), false, (b), (short)0, (c), false, false)

// ---------------------------------------------------------------- fragment loaders
// A (16x32 bf16, MxK): lane<16 -> M=lane, K in {kb..kb+15}
//                      lane>=16 -> M=lane-16, K in {kb+16..kb+31}
__device__ __forceinline__ v16bf load_a_frag(const bf16* rowbase, int kb, int lane) {
  int klo = kb + ((lane >> 4) << 4);
  return *(const v16bf*)(rowbase + klo);
}
// B (32x16 bf16, KxN) with W stored row-major [n][k]: per-lane 16 contiguous k.
__device__ __forceinline__ v16bf load_b_frag(const bf16* W, int ldb, int ntile,
                                             int kb, int lane) {
  int n = (ntile << 4) + (lane & 15);
  int k = kb + ((lane >> 4) << 4);
  return *(const v16bf*)(W + (size_t)n * ldb + k);
}

// ---------------------------------------------------------------- weight conversion
// Wih [NL][2][1636][818] -> Wihb [NL*2][GP][KP] bf16, gate-padded rows
__global__ __launch_bounds__(256) void cvt_wih(const float* __restrict__ src,
                                               bf16* __restrict__ dst, long total) {
  for (long i = (long)blockIdx.x * blockDim.x + threadIdx.x; i < total;
       i += (long)gridDim.x * blockDim.x) {
    int k  = (int)(i % KP);
    long r0 = i / KP;
    int np = (int)(r0 % GP);
    int ld = (int)(r0 / GP);              // layer*2 + dir
    int g = np / HP, r = np - g * HP;
    float v = 0.0f;
    if (r < HID && k < 818)
      v = src[((long)ld * 1636 + g * HID + r) * 818 + k];
    dst[i] = (bf16)v;
  }
}
// Whh [NL][2][1636][409] -> Whhb [NL*2][GP][HP]
__global__ __launch_bounds__(256) void cvt_whh(const float* __restrict__ src,
                                               bf16* __restrict__ dst, long total) {
  for (long i = (long)blockIdx.x * blockDim.x + threadIdx.x; i < total;
       i += (long)gridDim.x * blockDim.x) {
    int k  = (int)(i % HP);
    long r0 = i / HP;
    int np = (int)(r0 % GP);
    int ld = (int)(r0 / GP);
    int g = np / HP, r = np - g * HP;
    float v = 0.0f;
    if (r < HID && k < HID)
      v = src[((long)ld * 1636 + g * HID + r) * HID + k];
    dst[i] = (bf16)v;
  }
}
// bsum [NL*2][GP] = bih + bhh (gate-padded, fp32)
__global__ __launch_bounds__(256) void cvt_bsum(const float* __restrict__ bih,
                                                const float* __restrict__ bhh,
                                                float* __restrict__ dst, int total) {
  for (int i = blockIdx.x * blockDim.x + threadIdx.x; i < total;
       i += gridDim.x * blockDim.x) {
    int np = i % GP, ld = i / GP;
    int g = np / HP, r = np - g * HP;
    float v = 0.0f;
    if (r < HID) {
      long s = (long)ld * 1636 + g * HID + r;
      v = bih[s] + bhh[s];
    }
    dst[i] = v;
  }
}
// W1 [512][818] -> W1b [512][KP]
__global__ __launch_bounds__(256) void cvt_w1(const float* __restrict__ src,
                                              bf16* __restrict__ dst, int total) {
  for (int i = blockIdx.x * blockDim.x + threadIdx.x; i < total;
       i += gridDim.x * blockDim.x) {
    int k = i % KP, n = i / KP;
    dst[i] = (bf16)(k < 818 ? src[n * 818 + k] : 0.0f);
  }
}

// ---------------------------------------------------------------- front-end
// x0[t*B+b][0:768]=enc, [768:818]=ngram@Wprojᵀ, [818:832]=0   (bf16)
__global__ __launch_bounds__(256) void build_x0(const float* __restrict__ enc,
                                                const float* __restrict__ Wproj,
                                                bf16* __restrict__ xin) {
  __shared__ float ng[D_];
  int row = blockIdx.x;                 // t*B + b
  int t = row >> 5, b = row & 31;
  const float* e = enc + ((size_t)b * T_ + t) * D_;
  bf16* xr = xin + (size_t)row * KP;
  for (int k = threadIdx.x; k < D_; k += 256) {
    float v = e[k];
    float n;
    if (t == T_ - 1)    n = v;                                  // last token = itself
    else if (t == 0)    n = 0.5f * (v + e[D_ + k]);             // mean of [0:2]
    else                n = (e[k - D_] + v + e[D_ + k]) * (1.0f / 3.0f);
    ng[k] = n;
    xr[k] = (bf16)v;
  }
  __syncthreads();
  int j = threadIdx.x;
  if (j < 50) {
    const float* wp = Wproj + (size_t)j * D_;
    float s = 0.0f;
    for (int k = 0; k < D_; ++k) s += ng[k] * wp[k];
    xr[D_ + j] = (bf16)s;
  } else if (j < 64) {
    xr[D_ + j] = (bf16)0.0f;            // pad cols 818..831
  }
}

// ---------------------------------------------------------------- WMMA GEMM
// C[m][n] = sum_k A[m,k]*W[n,k] + bias[n]; one wave = 16x64 tile, K-loop of 32.
// grid.y selects a direction slice of W / bias / C.
__global__ __launch_bounds__(256)
void gemm_xw_bf16(const bf16* __restrict__ A, int lda,
                  const bf16* __restrict__ W, int ldb, size_t wStride,
                  const float* __restrict__ bias, int biasStride,
                  float* __restrict__ C, int ldc, size_t cStride,
                  int K, int Mtiles, int nGroups) {
  int lane = threadIdx.x & 31;
  int wid  = blockIdx.x * 8 + (threadIdx.x >> 5);
  if (wid >= Mtiles * nGroups) return;
  int mt   = wid / nGroups;
  int ngrp = wid - mt * nGroups;
  const bf16*  Wd = W + blockIdx.y * wStride;
  const float* bd = bias ? bias + (size_t)blockIdx.y * biasStride : nullptr;
  float*       Cd = C + blockIdx.y * cStride;

  const bf16* arow = A + (size_t)(mt * 16 + (lane & 15)) * lda;
  int n0 = ngrp * 4;
  v8f acc[4] = {};
  for (int kb = 0; kb < K; kb += 32) {
    __builtin_prefetch(arow + kb + 128, 0, 1);    // global_prefetch_b8
    v16bf a = load_a_frag(arow, kb, lane);
#pragma unroll
    for (int j = 0; j < 4; ++j) {
      v16bf b = load_b_frag(Wd, ldb, n0 + j, kb, lane);
      acc[j] = WMMA_BF16(a, b, acc[j]);
    }
  }
  int mb = (lane >> 4) << 3, col = lane & 15;
#pragma unroll
  for (int j = 0; j < 4; ++j) {
    int n = (n0 + j) * 16 + col;
    float bv = bd ? bd[n] : 0.0f;
#pragma unroll
    for (int v = 0; v < 8; ++v)
      Cd[(size_t)(mt * 16 + mb + v) * ldc + n] = acc[j][v] + bv;
  }
}

// ---------------------------------------------------------------- LSTM recurrence
// One persistent workgroup per direction. 26 waves, each owns 4 n-tiles x 2 m-tiles.
// h_prev (bf16) in LDS; gates staged through LDS; cell state in registers.
__global__ __launch_bounds__(832, 1)
void lstm_recur(const bf16* __restrict__ Whh,   // [2][GP][HP]
                const float* __restrict__ pre,  // [2][TB][GP]
                bf16* __restrict__ xout) {      // [TB][KP] (zeroed by host memset)
  __shared__ __align__(16) bf16 hbuf[B_ * HP];  //  26.6 KB
  __shared__ __align__(16) bf16 gbuf[B_ * GP];  // 106.5 KB
  const int dir  = blockIdx.x;
  const int tid  = threadIdx.x;
  const int lane = tid & 31;
  const int wave = tid >> 5;                    // 0..25
  const bf16*  Wd   = Whh + (size_t)dir * GP * HP;
  const float* pred = pre + (size_t)dir * TB * GP;

  for (int i = tid; i < B_ * HP; i += 832) hbuf[i] = (bf16)0.0f;
  float c[16];
#pragma unroll
  for (int j = 0; j < 16; ++j) c[j] = 0.0f;
  __syncthreads();

  const bf16* arow0 = hbuf + (size_t)(lane & 15) * HP;         // rows 0..15
  const bf16* arow1 = hbuf + (size_t)(16 + (lane & 15)) * HP;  // rows 16..31
  const int n0 = wave * 4;
  const int mb = (lane >> 4) << 3, col = lane & 15;

  for (int s = 0; s < T_; ++s) {
    const int t = dir ? (T_ - 1 - s) : s;

    // ---- gates += h_prev @ Whhᵀ  (WMMA, A from LDS)
    v8f acc[2][4] = {};
#pragma unroll 1
    for (int kt = 0; kt < 13; ++kt) {
      int kb = kt * 32;
      v16bf a0 = load_a_frag(arow0, kb, lane);
      v16bf a1 = load_a_frag(arow1, kb, lane);
#pragma unroll
      for (int j = 0; j < 4; ++j) {
        v16bf b = load_b_frag(Wd, HP, n0 + j, kb, lane);
        acc[0][j] = WMMA_BF16(a0, b, acc[0][j]);
        acc[1][j] = WMMA_BF16(a1, b, acc[1][j]);
      }
    }
#pragma unroll
    for (int mt = 0; mt < 2; ++mt)
#pragma unroll
      for (int j = 0; j < 4; ++j)
#pragma unroll
        for (int v = 0; v < 8; ++v)
          gbuf[(size_t)(mt * 16 + mb + v) * GP + (n0 + j) * 16 + col] =
              (bf16)acc[mt][j][v];
    __syncthreads();

    // ---- cell update: 13312 cells over 832 threads (16 each)
    const float* prow = pred + (size_t)t * B_ * GP;
#pragma unroll
    for (int j = 0; j < 16; ++j) {
      int flat = tid + j * 832;
      int b = flat / HP, r = flat - b * HP;
      const float* p = prow + (size_t)b * GP;
      const bf16*  g = gbuf + (size_t)b * GP;
      float gi = (float)g[r]          + p[r];
      float gf = (float)g[HP + r]     + p[HP + r];
      float gg = (float)g[2 * HP + r] + p[2 * HP + r];
      float go = (float)g[3 * HP + r] + p[3 * HP + r];
      float si = 1.0f / (1.0f + __expf(-gi));
      float sf = 1.0f / (1.0f + __expf(-gf));
      float so = 1.0f / (1.0f + __expf(-go));
      float cn = sf * c[j] + si * tanhf(gg);
      c[j] = cn;
      float h = so * tanhf(cn);
      hbuf[b * HP + r] = (bf16)h;
      if (r < HID)
        xout[(size_t)(t * B_ + b) * KP + (dir ? HID + r : r)] = (bf16)h;
    }
    __syncthreads();
  }
}

// ---------------------------------------------------------------- FC2 head
__global__ __launch_bounds__(32) void fc2_kernel(const float* __restrict__ fch,
                                                 const float* __restrict__ W2,
                                                 const float* __restrict__ b2,
                                                 float* __restrict__ out) {
  int row = blockIdx.x, v = threadIdx.x;
  if (v >= VOC) return;
  const float* f = fch + (size_t)row * 512;
  const float* w = W2 + (size_t)v * 512;
  float s = b2[v];
  for (int k = 0; k < 512; ++k) s += f[k] * w[k];
  int b = row & 31, t = row >> 5;
  out[((size_t)b * T_ + t) * VOC + v] = s;
}

// ---------------------------------------------------------------- launch
extern "C" void kernel_launch(void* const* d_in, const int* in_sizes, int n_in,
                              void* d_out, int out_size, void* d_ws, size_t ws_size,
                              hipStream_t stream) {
  const float* enc   = (const float*)d_in[0];
  const float* Wproj = (const float*)d_in[1];
  const float* Wih   = (const float*)d_in[2];
  const float* Whh   = (const float*)d_in[3];
  const float* bih   = (const float*)d_in[4];
  const float* bhh   = (const float*)d_in[5];
  const float* W1    = (const float*)d_in[6];
  const float* b1    = (const float*)d_in[7];
  const float* W2    = (const float*)d_in[8];
  const float* b2    = (const float*)d_in[9];
  float* out = (float*)d_out;

  char* w = (char*)d_ws;
  size_t off = 0;
  auto take = [&](size_t bytes) {
    char* p = w + off;
    off = (off + bytes + 255) & ~(size_t)255;
    return p;
  };
  bf16*  xA   = (bf16*)take((size_t)TB * KP * 2);
  bf16*  xB   = (bf16*)take((size_t)TB * KP * 2);
  float* pre  = (float*)take((size_t)2 * TB * GP * 4);
  bf16*  Wihb = (bf16*)take((size_t)NL_ * 2 * GP * KP * 2);
  bf16*  Whhb = (bf16*)take((size_t)NL_ * 2 * GP * HP * 2);
  float* bsum = (float*)take((size_t)NL_ * 2 * GP * 4);
  bf16*  W1b  = (bf16*)take((size_t)512 * KP * 2);
  float* fch  = (float*)take((size_t)TB * 512 * 4);

  // weight conversion (deterministic every call)
  {
    long n = (long)NL_ * 2 * GP * KP;
    cvt_wih<<<(unsigned)((n + 255) / 256), 256, 0, stream>>>(Wih, Wihb, n);
  }
  {
    long n = (long)NL_ * 2 * GP * HP;
    cvt_whh<<<(unsigned)((n + 255) / 256), 256, 0, stream>>>(Whh, Whhb, n);
  }
  cvt_bsum<<<(NL_ * 2 * GP + 255) / 256, 256, 0, stream>>>(bih, bhh, bsum,
                                                           NL_ * 2 * GP);
  cvt_w1<<<(512 * KP + 255) / 256, 256, 0, stream>>>(W1, W1b, 512 * KP);

  // front end: ngram + projection + concat -> bf16 x0
  build_x0<<<TB, 256, 0, stream>>>(enc, Wproj, xA);

  bf16* ping = xA;
  bf16* pong = xB;
  const int Mtiles = TB / 16;                 // 1024
  for (int l = 0; l < NL_; ++l) {
    // pre[dir] = x @ Wihᵀ + (bih+bhh), both directions via grid.y
    int nGroups = GP / 64;                    // 26
    int jobs = Mtiles * nGroups;
    dim3 g((jobs + 7) / 8, 2);
    gemm_xw_bf16<<<g, 256, 0, stream>>>(
        ping, KP, Wihb + (size_t)l * 2 * GP * KP, KP, (size_t)GP * KP,
        bsum + (size_t)l * 2 * GP, GP, pre, GP, (size_t)TB * GP, KP, Mtiles,
        nGroups);
    hipMemsetAsync(pong, 0, (size_t)TB * KP * 2, stream);
    lstm_recur<<<2, 832, 0, stream>>>(Whhb + (size_t)l * 2 * GP * HP, pre, pong);
    bf16* t = ping; ping = pong; pong = t;
  }

  // FC1: fch = out @ W1ᵀ + b1   (WMMA)
  {
    int nGroups = 512 / 64;                   // 8
    int jobs = Mtiles * nGroups;
    dim3 g((jobs + 7) / 8, 1);
    gemm_xw_bf16<<<g, 256, 0, stream>>>(ping, KP, W1b, KP, 0, b1, 0, fch, 512, 0,
                                        KP, Mtiles, nGroups);
  }
  // FC2: logits
  fc2_kernel<<<TB, 32, 0, stream>>>(fch, W2, b2, out);
}
